// CustomRNN_30520037605920
// MI455X (gfx1250) — compile-verified, matrix operand order checked by reading, not run on hardware
//
#include <hip/hip_runtime.h>
#include <hip/hip_bf16.h>

typedef __bf16 bf16;
typedef bf16 v16bf __attribute__((ext_vector_type(16)));
typedef bf16 v8bf  __attribute__((ext_vector_type(8)));
typedef float v8f  __attribute__((ext_vector_type(8)));

#define HID   1024
#define GATES 4096
#define BATCH 64
#define SEQ   512
#define VOC   256

__device__ __forceinline__ float sigf(float x) { return 1.f / (1.f + __expf(-x)); }
__device__ __forceinline__ float tanhx(float x) {
    x = fminf(fmaxf(x, -15.f), 15.f);
    float e = __expf(2.f * x);
    return (e - 1.f) / (e + 1.f);
}

// WGP-scope prefetch (scope field = 0): pulls the line into ALL cache levels,
// including the WGP cache. The __builtin_prefetch path can only reach
// SE/DEV/SYS scopes, which stop at GL2 and don't hide L2->L0 latency.
// TH=0 -> speculative: out-of-range addresses are silently dropped.
__device__ __forceinline__ void prefetch_wgp(const void* p) {
    asm volatile("global_prefetch_b8 %0, off" :: "v"((unsigned long long)(uintptr_t)p));
}

// Pack fp32 weights [K x 4096] (rows 0..K0-1 from Wa, rest from Wb) into bf16
// WMMA B-fragment order: [j16][kt][gate][lane][elem16], lane = column (lane&15),
// elem order matches 16-bit operand layout (K 0-7 & 16-23 for lanes 0-15, etc.)
__global__ void pack_weights(const float* __restrict__ Wa,
                             const float* __restrict__ Wb,
                             int K0, long long total, bf16* __restrict__ out, int KT)
{
    long long i = (long long)blockIdx.x * blockDim.x + threadIdx.x;
    if (i >= total) return;
    int elem = (int)(i & 15);
    long long r = i >> 4;
    int lane = (int)(r & 31); r >>= 5;
    int gate = (int)(r & 3);  r >>= 2;
    int kt   = (int)(r % KT);
    int j16  = (int)(r / KT);
    int kbase = (lane >> 4) * 8;
    int koff = (elem < 8) ? (kbase + elem) : (kbase + 16 + (elem - 8));
    int k = kt * 32 + koff;
    int col = gate * HID + j16 * 16 + (lane & 15);
    float v = (k < K0) ? Wa[(long long)k * GATES + col]
                       : Wb[(long long)(k - K0) * GATES + col];
    out[i] = (bf16)v;
}

// x [B,T,V] fp32 -> time-major bf16 [T,B,V]
__global__ void convert_x(const float* __restrict__ x, bf16* __restrict__ xbf)
{
    long long i = (long long)blockIdx.x * blockDim.x + threadIdx.x;
    if (i >= (long long)BATCH * SEQ * VOC) return;
    int v = (int)(i & (VOC - 1));
    long long r = i >> 8;
    int t = (int)(r & (SEQ - 1));
    int b = (int)(r >> 9);
    xbf[((long long)t * BATCH + b) * VOC + v] = (bf16)x[i];
}

__global__ void init_state(const float* __restrict__ h_in, const float* __restrict__ c_in,
                           float* __restrict__ h_state, float* __restrict__ c_state,
                           bf16* __restrict__ hbf0, bf16* __restrict__ hbf1)
{
    int i = blockIdx.x * blockDim.x + threadIdx.x;
    if (i >= 2 * BATCH * HID) return;
    h_state[i] = h_in[i];
    c_state[i] = c_in[i];
    int l = i >> 16;
    int idx = i & 65535;
    if (l == 0) hbf0[idx] = (bf16)h_in[i];
    else        hbf1[idx] = (bf16)h_in[i];
}

// One fused LSTM layer step. Grid (64 j-tiles, 2 row halves), 32 threads (1 wave).
// Each wave: 32 batch rows x (16 hidden units x 4 gates): per k-tile it loads
// 4 shared B fragments + 2 A fragments and issues 8 v_wmma_f32_16x16x32_bf16,
// then does the fused gate/cell update for its rows in-registers.
__global__ __launch_bounds__(32) void lstm_step(
    const bf16* __restrict__ a0, int ld0, int kt0,   // input activations (bf16, row-major)
    const bf16* __restrict__ a1, int kt1,            // h_prev (bf16, ld=HID)
    const bf16* __restrict__ Wp,                     // packed weights
    const float* __restrict__ bias,
    float* __restrict__ c_state,
    float* __restrict__ h_state,
    bf16*  __restrict__ hbf_out,
    float* __restrict__ outs, int t)
{
    const int KT = kt0 + kt1;
    const int j16  = blockIdx.x;
    const int rh   = blockIdx.y;                     // rows [32*rh, 32*rh+32)
    const int lane = threadIdx.x & 31;
    const int lrow  = lane & 15;
    const int khalf = (lane >> 4) * 8;

    v8f a0i = {}, a0f = {}, a0g = {}, a0o = {};      // rows 32*rh + 0..15
    v8f a1i = {}, a1f = {}, a1g = {}, a1o = {};      // rows 32*rh + 16..31
    const bf16* wbase = Wp + ((long long)j16 * KT) * 2048;   // this WG's weight stream
    const bf16* wptr  = wbase + lane * 16;
    // Prefetch stream: each lane grabs one 128B line; 32 lanes cover the full
    // 4KB k-tile block, issued 4 tiles (16KB) ahead of the demand loads.
    const bf16* pfptr = wbase + 4 * 2048 + lane * 64;

    auto ldA = [&](const bf16* pa) {
        v8bf lo = *(const v8bf*)pa;                  // K = khalf + 0..7
        v8bf hi = *(const v8bf*)(pa + 16);           // K = khalf + 16..23
        return __builtin_shufflevector(lo, hi,
            0, 1, 2, 3, 4, 5, 6, 7, 8, 9, 10, 11, 12, 13, 14, 15);
    };

    auto mma = [&](const bf16* paA, long long ldA_, long long kt) {
        const bf16* pw = wptr + kt * 2048;
        v16bf aA = ldA(paA);
        v16bf aB = ldA(paA + 16 * ldA_);
        v16bf b0 = *(const v16bf*)(pw);
        v16bf b1 = *(const v16bf*)(pw + 512);
        v16bf b2 = *(const v16bf*)(pw + 1024);
        v16bf b3 = *(const v16bf*)(pw + 1536);
        prefetch_wgp(pfptr + kt * 2048);             // k-tile kt+4, full 4KB coverage
        a0i = __builtin_amdgcn_wmma_f32_16x16x32_bf16(false, aA, false, b0, (short)0, a0i, false, false);
        a0f = __builtin_amdgcn_wmma_f32_16x16x32_bf16(false, aA, false, b1, (short)0, a0f, false, false);
        a0g = __builtin_amdgcn_wmma_f32_16x16x32_bf16(false, aA, false, b2, (short)0, a0g, false, false);
        a0o = __builtin_amdgcn_wmma_f32_16x16x32_bf16(false, aA, false, b3, (short)0, a0o, false, false);
        a1i = __builtin_amdgcn_wmma_f32_16x16x32_bf16(false, aB, false, b0, (short)0, a1i, false, false);
        a1f = __builtin_amdgcn_wmma_f32_16x16x32_bf16(false, aB, false, b1, (short)0, a1f, false, false);
        a1g = __builtin_amdgcn_wmma_f32_16x16x32_bf16(false, aB, false, b2, (short)0, a1g, false, false);
        a1o = __builtin_amdgcn_wmma_f32_16x16x32_bf16(false, aB, false, b3, (short)0, a1o, false, false);
    };

    const int rowA = rh * 32 + lrow;
    for (int kt = 0; kt < kt0; ++kt)
        mma(a0 + (long long)rowA * ld0 + kt * 32 + khalf, ld0, kt);
    for (int kt = 0; kt < kt1; ++kt)
        mma(a1 + (long long)rowA * HID + kt * 32 + khalf, HID, kt0 + kt);

    // Fused pointwise: this wave owns all 4 gates for hidden units j16*16..+15,
    // rows 32*rh..+31.
    const int j = j16 * 16 + lrow;
    const float bi  = bias[j];
    const float bf_ = bias[HID + j];
    const float bg  = bias[2 * HID + j];
    const float bo  = bias[3 * HID + j];

    auto cell = [&](v8f& zi8, v8f& zf8, v8f& zg8, v8f& zo8, int rtop) {
        #pragma unroll
        for (int v = 0; v < 8; ++v) {
            int r = rtop + v;
            long long idx = (long long)r * HID + j;
            float zi = zi8[v] + bi;
            float zf = zf8[v] + bf_;
            float zg = zg8[v] + bg;
            float zo = zo8[v] + bo;
            float cp = c_state[idx];
            float cn = sigf(zf) * cp + sigf(zi) * tanhx(zg);
            float hn = sigf(zo) * tanhx(cn);
            c_state[idx] = cn;
            h_state[idx] = hn;
            hbf_out[idx] = (bf16)hn;
            if (outs) outs[((long long)r * SEQ + t) * HID + j] = hn;
        }
    };
    const int rsub = (lane >> 4) ? 8 : 0;
    cell(a0i, a0f, a0g, a0o, rh * 32 + rsub);
    cell(a1i, a1f, a1g, a1o, rh * 32 + 16 + rsub);
}

extern "C" void kernel_launch(void* const* d_in, const int* in_sizes, int n_in,
                              void* d_out, int out_size, void* d_ws, size_t ws_size,
                              hipStream_t stream)
{
    const float* x     = (const float*)d_in[0];
    const float* h_in  = (const float*)d_in[1];
    const float* c_in  = (const float*)d_in[2];
    const float* W_ih0 = (const float*)d_in[3];
    const float* W_hh0 = (const float*)d_in[4];
    const float* b0    = (const float*)d_in[5];
    const float* W_ih1 = (const float*)d_in[6];
    const float* W_hh1 = (const float*)d_in[7];
    const float* b1    = (const float*)d_in[8];

    char* ws = (char*)d_ws;
    size_t off = 0;
    auto carve = [&](size_t bytes) {
        void* p = ws + off;
        off = (off + bytes + 255) & ~(size_t)255;
        return p;
    };
    bf16* pW0 = (bf16*)carve((size_t)1280 * 4096 * 2);   // L0 packed weights (bf16)
    bf16* pW1 = (bf16*)carve((size_t)2048 * 4096 * 2);   // L1 packed weights
    bf16* xbf = (bf16*)carve((size_t)SEQ * BATCH * VOC * 2);
    bf16* hbf0[2] = { (bf16*)carve(BATCH * HID * 2), (bf16*)carve(BATCH * HID * 2) };
    bf16* hbf1[2] = { (bf16*)carve(BATCH * HID * 2), (bf16*)carve(BATCH * HID * 2) };

    float* out = (float*)d_out;
    const long long OUTS = (long long)BATCH * SEQ * HID;
    float* h_state = out + OUTS;                       // h_f [2,B,H] lives here
    float* c_state = out + OUTS + 2 * BATCH * HID;     // c_f [2,B,H]

    // One-time prologue (inside the graph; deterministic every call).
    long long tot0 = (long long)1280 * 4096;
    pack_weights<<<(unsigned)((tot0 + 255) / 256), 256, 0, stream>>>(W_ih0, W_hh0, VOC, tot0, pW0, 40);
    long long tot1 = (long long)2048 * 4096;
    pack_weights<<<(unsigned)((tot1 + 255) / 256), 256, 0, stream>>>(W_ih1, W_hh1, HID, tot1, pW1, 64);
    long long tx = (long long)BATCH * SEQ * VOC;
    convert_x<<<(unsigned)((tx + 255) / 256), 256, 0, stream>>>(x, xbf);
    init_state<<<(2 * BATCH * HID + 255) / 256, 256, 0, stream>>>(h_in, c_in, h_state, c_state,
                                                                  hbf0[0], hbf1[0]);

    dim3 grid(64, 2), block(32);
    for (int t = 0; t < SEQ; ++t) {
        int rp = t & 1, wp = (t + 1) & 1;
        // layer 0: A = [x_t (256) | h0_prev (1024)]
        lstm_step<<<grid, block, 0, stream>>>(
            xbf + (long long)t * BATCH * VOC, VOC, 8,
            hbf0[rp], 32, pW0, b0,
            c_state, h_state, hbf0[wp],
            (float*)nullptr, t);
        // layer 1: A = [h0_new (1024) | h1_prev (1024)]
        lstm_step<<<grid, block, 0, stream>>>(
            hbf0[wp], HID, 32,
            hbf1[rp], 32, pW1, b1,
            c_state + BATCH * HID, h_state + BATCH * HID, hbf1[wp],
            out, t);
    }
    (void)in_sizes; (void)n_in; (void)out_size; (void)ws_size;
}
